// MoEHead_31550829756913
// MI455X (gfx1250) — matrix operand order repaired and checked by reading, not compiled, not run on hardware
//
#include <hip/hip_runtime.h>

// ---------------- problem constants ----------------
#define TOK   8192      // T = B*N
#define DIM   1024      // D
#define NEXP  8         // E
#define TOPK  2         // K
#define HID   4096      // H
#define OOUT  1024      // O
#define INV_TAU (1.0f/1.5f)

#define MT    64        // tokens per tile (4 WMMA row-tiles)
#define TILES (TOK/MT)  // 128 tiles per expert (worst case)
#define KP    40        // padded K-stride (elements) for col-major weight slabs (80B, 16B-aligned)

typedef __attribute__((ext_vector_type(16))) __bf16 bf16x16;
typedef __attribute__((ext_vector_type(8)))  __bf16 bf16x8;
typedef __attribute__((ext_vector_type(8)))  float  f32x8;

union Frag {
    bf16x16 v;
    bf16x8  h[2];
    __bf16  e[16];
};

// Split workgroup barrier that drains ONLY DScnt (LDS producer/consumer
// handoff). Unlike __syncthreads(), it does NOT force outstanding global
// loads (LOADcnt) to complete, so the register-pipelined weight prefetch
// stays in flight across stage barriers.
__device__ __forceinline__ void lds_barrier() {
    asm volatile(
        "s_wait_dscnt 0x0\n\t"
        "s_barrier_signal -1\n\t"
        "s_barrier_wait -1" ::: "memory");
}

// ---------------- workspace layout (bytes) ----------------
#define WS_COUNTS 0
#define WS_GATES  256
#define WS_EIDX   (WS_GATES + TOK*TOPK*4)
#define WS_LOGITS (WS_EIDX  + TOK*TOPK*4)
#define WS_ELIST  (WS_LOGITS + TOK*NEXP*4)

// =====================================================================
// Router: one wave32 per token.
// =====================================================================
__global__ __launch_bounds__(256) void router_kernel(
    const float* __restrict__ h, const float* __restrict__ rw,
    const float* __restrict__ rb,
    float* __restrict__ gates, int* __restrict__ eidx,
    float* __restrict__ logits_out, int* __restrict__ counts,
    int* __restrict__ elist)
{
    const int lane = threadIdx.x & 31;
    const int wave = threadIdx.x >> 5;
    const int t    = blockIdx.x * 8 + wave;

    float acc[NEXP];
#pragma unroll
    for (int e = 0; e < NEXP; ++e) acc[e] = 0.f;

    const float* hrow = h + (size_t)t * DIM;
    const float4* rw4 = (const float4*)rw;  // rw is [D][E], E=8 -> 2 float4 per row
    for (int d = lane; d < DIM; d += 32) {
        float  hv = hrow[d];
        float4 a  = rw4[d * 2 + 0];
        float4 b  = rw4[d * 2 + 1];
        acc[0] += hv * a.x; acc[1] += hv * a.y; acc[2] += hv * a.z; acc[3] += hv * a.w;
        acc[4] += hv * b.x; acc[5] += hv * b.y; acc[6] += hv * b.z; acc[7] += hv * b.w;
    }
#pragma unroll
    for (int e = 0; e < NEXP; ++e)
#pragma unroll
        for (int off = 16; off > 0; off >>= 1)
            acc[e] += __shfl_xor(acc[e], off, 32);

    if (lane == 0) {
        float l[NEXP];
#pragma unroll
        for (int e = 0; e < NEXP; ++e) {
            l[e] = (acc[e] + rb[e]) * INV_TAU;
            logits_out[t * NEXP + e] = l[e];
        }
        int i0 = 0;
#pragma unroll
        for (int e = 1; e < NEXP; ++e) if (l[e] > l[i0]) i0 = e;
        int i1 = (i0 == 0) ? 1 : 0;
#pragma unroll
        for (int e = 0; e < NEXP; ++e) if (e != i0 && l[e] > l[i1]) i1 = e;
        float ex = __expf(l[i1] - l[i0]);
        float g0 = 1.f / (1.f + ex);
        float g1 = ex  / (1.f + ex);
        gates[t * 2 + 0] = g0;  gates[t * 2 + 1] = g1;
        eidx [t * 2 + 0] = i0;  eidx [t * 2 + 1] = i1;
        int p0 = atomicAdd(&counts[i0], 1); elist[i0 * TOK + p0] = t * 2 + 0;
        int p1 = atomicAdd(&counts[i1], 1); elist[i1 * TOK + p1] = t * 2 + 1;
    }
}

// =====================================================================
// Pipelined staging helpers: global loads into registers (issued early),
// convert + 2x ds_store_b128 later, after the consumer barrier.
// =====================================================================
__device__ __forceinline__ void load16(const float* __restrict__ g, size_t stride,
                                       float v[16]) {
#pragma unroll
    for (int i = 0; i < 16; ++i) v[i] = g[(size_t)i * stride];
}

__device__ __forceinline__ void store16(const float v[16], __bf16* dst) {
    bf16x8 p0, p1;
#pragma unroll
    for (int i = 0; i < 8; ++i) { p0[i] = (__bf16)v[i]; p1[i] = (__bf16)v[8 + i]; }
    ((bf16x8*)dst)[0] = p0;
    ((bf16x8*)dst)[1] = p1;
}

// =====================================================================
// Expert MLP tile: 64 tokens x full O for one expert, 8 waves, 256 thr.
//   GEMM1: hid = silu(X[64,1024] @ w1 + b1) in 128-col chunks (bf16 WMMA)
//   GEMM2: Y[64,1024] += hid_chunk @ w2 (bf16 WMMA, f32 acc: 256 VGPR/wave)
// Weight slabs col-major in LDS (B-frags = 2x ds_load_b128); staging is
// software-pipelined in registers; split barriers drain DScnt only.
// =====================================================================
__global__ __launch_bounds__(256)
__attribute__((amdgpu_waves_per_eu(2, 2)))
void expert_mlp_kernel(
    const float* __restrict__ hmat, const float* __restrict__ w1,
    const float* __restrict__ b1,   const float* __restrict__ w2,
    const float* __restrict__ b2,   const float* __restrict__ gates,
    const int* __restrict__ counts, const int* __restrict__ elist,
    float* __restrict__ y)
{
    const int e     = blockIdx.x / TILES;
    const int tile  = blockIdx.x % TILES;
    const int cnt   = counts[e];
    const int start = tile * MT;
    if (start >= cnt) return;

    extern __shared__ char smem[];
    __bf16* Xs  = (__bf16*)smem;                 // [64][1024] row-major   128KB
    __bf16* Hs  = Xs  + MT * DIM;                // [64][128]  row-major    16KB
    __bf16* W1s = Hs  + MT * 128;                // [128][KP]  col-major    10KB
    __bf16* W2s = W1s + 128 * KP;                // [1024][KP] col-major    80KB
    int*    rowPair = (int*)(W2s + OOUT * KP);
    float*  rowGate = (float*)(rowPair + MT);
    int*    rowTok  = (int*)(rowGate + MT);

    const int tid   = threadIdx.x;
    const int lane  = tid & 31;
    const int wave  = tid >> 5;
    const int n15   = lane & 15;
    const int koffA = (lane >= 16) ? 8  : 0;   // 16-bit A-frag K offset (ISA 7.12.2)
    const int koffB = (lane >= 16) ? 8  : 0;   // 16-bit B-frag K offset
    const int mhi   = (lane >= 16) ? 8  : 0;   // C/D row offset

    if (tid < MT) {
        int pair = (start + tid < cnt) ? elist[e * TOK + start + tid] : -1;
        rowPair[tid] = pair;
        rowTok [tid] = (pair >= 0) ? (pair >> 1) : 0;
        rowGate[tid] = (pair >= 0) ? gates[pair] : 0.f;
    }
    lds_barrier();

    // ---- stage gathered token rows as bf16 (8 floats per b128 store) ----
#pragma unroll
    for (int u = 0; u < (MT * DIM) / (256 * 8); ++u) {   // 32 units/thread
        int id = u * 256 + tid;
        int r  = id >> 7;               // 128 groups of 8 per row
        int d8 = (id & 127) << 3;
        float4 a = make_float4(0.f, 0.f, 0.f, 0.f), b4 = a;
        if (rowPair[r] >= 0) {
            const float* src = hmat + (size_t)rowTok[r] * DIM + d8;
            a  = ((const float4*)src)[0];
            b4 = ((const float4*)src)[1];
        }
        bf16x8 p;
        p[0] = (__bf16)a.x;  p[1] = (__bf16)a.y;  p[2] = (__bf16)a.z;  p[3] = (__bf16)a.w;
        p[4] = (__bf16)b4.x; p[5] = (__bf16)b4.y; p[6] = (__bf16)b4.z; p[7] = (__bf16)b4.w;
        *(bf16x8*)&Xs[r * DIM + d8] = p;
    }
    lds_barrier();

    // per-thread staging coordinates
    const int c1  = tid & 127;                 // w1 slab: 32(K) x 128(cols), 1 unit
    const int kh1 = (tid >> 7) << 4;
    __bf16* const dst1 = &W1s[c1 * KP + kh1];
    const float* const w1base = w1 + (size_t)e * DIM * HID + (size_t)kh1 * HID + c1;

    int   c2[8], kh2[8];                       // w2 slab: 32(K) x 1024(cols), 8 units
    __bf16* dst2[8];
#pragma unroll
    for (int u = 0; u < 8; ++u) {
        int id = u * 256 + tid;
        c2[u]  = id & 1023;
        kh2[u] = (id >> 10) << 4;
        dst2[u] = &W2s[c2[u] * KP + kh2[u]];
    }
    const float* const w2base = w2 + (size_t)e * HID * OOUT;

    const f32x8 vz = {};
    f32x8 acc2[4][8];
#pragma unroll
    for (int rt = 0; rt < 4; ++rt)
#pragma unroll
        for (int s = 0; s < 8; ++s) acc2[rt][s] = vz;

    const int myCol = wave * 16 + n15;         // GEMM1 column owned by this lane

    // prime the pipelines: first w1 slab (h0=0,d0=0), first w2 slab (h0=0,kk=0)
    float v1[16];
    load16(w1base, (size_t)HID, v1);           // rows d0=0+kh1.., col h0+c1
    float v2[8][16];
#pragma unroll
    for (int u = 0; u < 8; ++u)
        load16(w2base + (size_t)kh2[u] * OOUT + c2[u], (size_t)OOUT, v2[u]);

    for (int h0 = 0; h0 < HID; h0 += 128) {
        // ---------------- GEMM1: acc1[64 x 128-chunk] over K = D ----------------
        f32x8 acc1[4] = {vz, vz, vz, vz};
        for (int d0 = 0; d0 < DIM; d0 += 32) {
            store16(v1, dst1);                 // commit pipelined slab
            lds_barrier();

            // issue next w1 slab's loads (overlaps with WMMA below; LOADcnt
            // stays outstanding across the split barriers)
            {
                int td = d0 + 32, th = h0;
                if (td == DIM) { td = 0; th = h0 + 128; }
                if (th < HID)
                    load16(w1base + (size_t)td * HID + th, (size_t)HID, v1);
            }

            Frag bf;
            bf.h[0] = *(const bf16x8*)&W1s[myCol * KP + koffB];
            bf.h[1] = *(const bf16x8*)&W1s[myCol * KP + koffB + 16];
            Frag af[4];
#pragma unroll
            for (int rt = 0; rt < 4; ++rt) {
                const int row = rt * 16 + n15;
                af[rt].h[0] = *(const bf16x8*)&Xs[row * DIM + d0 + koffA];
                af[rt].h[1] = *(const bf16x8*)&Xs[row * DIM + d0 + koffA + 16];
            }
#pragma unroll
            for (int rt = 0; rt < 4; ++rt)
                acc1[rt] = __builtin_amdgcn_wmma_f32_16x16x32_bf16(
                    false, af[rt].v, false, bf.v, (short)0, acc1[rt], false, false);
            lds_barrier();
        }

        // ---------------- bias + SiLU -> Hs (bf16) ----------------
        const float b1v = b1[e * HID + h0 + myCol];
#pragma unroll
        for (int rt = 0; rt < 4; ++rt)
#pragma unroll
            for (int r = 0; r < 8; ++r) {
                int m = rt * 16 + r + mhi;
                float v = acc1[rt][r] + b1v;
                Hs[m * 128 + myCol] = (__bf16)(v / (1.f + __expf(-v)));
            }

        // ---------------- GEMM2: acc2 += Hs[64 x 128] @ w2[128 x 1024] ----------------
        for (int kk = 0; kk < 128; kk += 32) {
#pragma unroll
            for (int u = 0; u < 8; ++u) store16(v2[u], dst2[u]);  // commit slab
            lds_barrier();

            // issue next w2 slab's loads (overlaps with WMMA below)
            {
                int tk = kk + 32, th = h0;
                if (tk == 128) { tk = 0; th = h0 + 128; }
                if (th < HID) {
#pragma unroll
                    for (int u = 0; u < 8; ++u)
                        load16(w2base + (size_t)(th + tk + kh2[u]) * OOUT + c2[u],
                               (size_t)OOUT, v2[u]);
                }
            }

            Frag af2[4];
#pragma unroll
            for (int rt = 0; rt < 4; ++rt) {
                const int row = rt * 16 + n15;
                af2[rt].h[0] = *(const bf16x8*)&Hs[row * 128 + kk + koffA];
                af2[rt].h[1] = *(const bf16x8*)&Hs[row * 128 + kk + koffA + 16];
            }
#pragma unroll
            for (int sub = 0; sub < 8; ++sub) {
                Frag bf2;
                const int c = wave * 128 + sub * 16 + n15;
                bf2.h[0] = *(const bf16x8*)&W2s[c * KP + koffB];
                bf2.h[1] = *(const bf16x8*)&W2s[c * KP + koffB + 16];
#pragma unroll
                for (int rt = 0; rt < 4; ++rt)
                    acc2[rt][sub] = __builtin_amdgcn_wmma_f32_16x16x32_bf16(
                        false, af2[rt].v, false, bf2.v, (short)0, acc2[rt][sub], false, false);
            }
            lds_barrier();
        }
    }

    // ---- epilogue: y[t] += gate * (acc + b2)  (exactly 2 adds/elem -> deterministic) ----
#pragma unroll
    for (int sub = 0; sub < 8; ++sub) {
        const int col = wave * 128 + sub * 16 + n15;
        const float b2v = b2[e * OOUT + col];
#pragma unroll
        for (int rt = 0; rt < 4; ++rt)
#pragma unroll
            for (int r = 0; r < 8; ++r) {
                int m = rt * 16 + r + mhi;
                int pair = rowPair[m];
                if (pair >= 0)
                    atomicAdd(&y[(size_t)rowTok[m] * OOUT + col],
                              rowGate[m] * (acc2[rt][sub][r] + b2v));
            }
    }
}

// =====================================================================
// Finalize: deterministic single-block reduction for aux/p_e/f_e/entropy.
// =====================================================================
__device__ float block_sum(float v, float* red) {
    int tid = threadIdx.x;
    red[tid] = v; __syncthreads();
    for (int off = 128; off > 0; off >>= 1) {
        if (tid < off) red[tid] += red[tid + off];
        __syncthreads();
    }
    float r = red[0]; __syncthreads();
    return r;
}

__global__ __launch_bounds__(256) void finalize_kernel(
    const float* __restrict__ logits, const float* __restrict__ gates,
    const int* __restrict__ eidx, float* __restrict__ outTail)
{
    __shared__ float red[256];
    const int tid = threadIdx.x;
    float psum[NEXP], lsum[NEXP], ent = 0.f;
#pragma unroll
    for (int e = 0; e < NEXP; ++e) { psum[e] = 0.f; lsum[e] = 0.f; }

    for (int t = tid; t < TOK; t += 256) {
        float l[NEXP], m = -3.4e38f;
#pragma unroll
        for (int e = 0; e < NEXP; ++e) { l[e] = logits[t * NEXP + e]; m = fmaxf(m, l[e]); }
        float s = 0.f, p[NEXP];
#pragma unroll
        for (int e = 0; e < NEXP; ++e) { p[e] = __expf(l[e] - m); s += p[e]; }
        float inv = 1.f / s, lse = __logf(s);
#pragma unroll
        for (int e = 0; e < NEXP; ++e) {
            float pe = p[e] * inv;
            psum[e] += pe;
            ent -= pe * ((l[e] - m) - lse);
        }
        int i0 = eidx[t * 2], i1 = eidx[t * 2 + 1];
        float g0 = gates[t * 2], g1 = gates[t * 2 + 1];
#pragma unroll
        for (int e = 0; e < NEXP; ++e)
            lsum[e] += (i0 == e ? g0 : 0.f) + (i1 == e ? g1 : 0.f);
    }

    float p_e[NEXP], f_e[NEXP];
    for (int e = 0; e < NEXP; ++e) p_e[e] = block_sum(psum[e], red) * (1.f / TOK);
    for (int e = 0; e < NEXP; ++e) f_e[e] = block_sum(lsum[e], red) * (1.f / TOK);
    float entropy = block_sum(ent, red) * (1.f / TOK);

    if (tid == 0) {
        float aux = 0.f;
#pragma unroll
        for (int e = 0; e < NEXP; ++e) aux += p_e[e] * f_e[e];
        aux = fmaxf((float)NEXP * aux - 1.f, 0.f);
        outTail[0] = aux;
#pragma unroll
        for (int e = 0; e < NEXP; ++e) outTail[1 + e] = p_e[e];
#pragma unroll
        for (int e = 0; e < NEXP; ++e) outTail[9 + e] = f_e[e];
        outTail[17] = entropy;
    }
}

// =====================================================================
extern "C" void kernel_launch(void* const* d_in, const int* in_sizes, int n_in,
                              void* d_out, int out_size, void* d_ws, size_t ws_size,
                              hipStream_t stream) {
    (void)in_sizes; (void)n_in; (void)out_size; (void)ws_size;
    const float* h  = (const float*)d_in[0];
    const float* rw = (const float*)d_in[1];
    const float* rb = (const float*)d_in[2];
    const float* w1 = (const float*)d_in[3];
    const float* b1 = (const float*)d_in[4];
    const float* w2 = (const float*)d_in[5];
    const float* b2 = (const float*)d_in[6];
    float* y = (float*)d_out;

    char* ws = (char*)d_ws;
    int*   counts = (int*)  (ws + WS_COUNTS);
    float* gates  = (float*)(ws + WS_GATES);
    int*   eidx   = (int*)  (ws + WS_EIDX);
    float* logits = (float*)(ws + WS_LOGITS);
    int*   elist  = (int*)  (ws + WS_ELIST);

    hipMemsetAsync(d_out, 0, (size_t)TOK * OOUT * sizeof(float), stream);
    hipMemsetAsync(counts, 0, 64, stream);

    router_kernel<<<TOK / 8, 256, 0, stream>>>(h, rw, rb, gates, eidx, logits, counts, elist);

    const size_t smem = (size_t)(MT * DIM + MT * 128 + 128 * KP + OOUT * KP) * sizeof(__bf16)
                        + MT * (4 + 4 + 4);
    (void)hipFuncSetAttribute((const void*)expert_mlp_kernel,
                              hipFuncAttributeMaxDynamicSharedMemorySize, (int)smem);
    expert_mlp_kernel<<<NEXP * TILES, 256, smem, stream>>>(
        h, w1, b1, w2, b2, gates, counts, elist, y);

    finalize_kernel<<<1, 256, 0, stream>>>(logits, gates, eidx, y + (size_t)TOK * OOUT);
}
